// KANNeuron_60705067762059
// MI455X (gfx1250) — matrix operand order repaired and checked by reading, not compiled, not run on hardware
//
#include <hip/hip_runtime.h>

// ---------------------------------------------------------------------------
// KAN layer for MI455X (gfx1250, wave32, WMMA).
//
// out = silu(x) @ BWᵀ + bspline(x) @ (SW*SC)ᵀ   ==  A(4096x9216) @ W(1024x9216)ᵀ
// with K = 1024*9 (per input feature: 1 silu value + 8 cubic B-spline bases).
//
// f16 operands / f32 accumulation via v_wmma_f32_16x16x32_f16.
// Workspace layout: A f16 [4096*9216] (75.5 MB), then W f16 [1024*9216] (18.9 MB).
// ---------------------------------------------------------------------------

typedef __attribute__((ext_vector_type(16))) _Float16 v16h;
typedef __attribute__((ext_vector_type(8)))  _Float16 v8h;
typedef __attribute__((ext_vector_type(8)))  float    v8f;

#define BATCH   4096
#define IN_DIM  1024
#define OUT_DIM 1024
#define NCOEF   8              // GRID_SIZE + SPLINE_ORDER
#define NKNOT   12             // arange(-3, 9)
#define KDIM    (IN_DIM * (NCOEF + 1))   // 9216
#define BK      32             // K per WMMA step (f16)

// --------------------------- Kernel 1: build A -----------------------------
// One thread per (batch m, input i): silu(x) and the 8 cubic B-spline bases.
// A is stored as 9 planes of IN_DIM per row: k = plane*IN_DIM + i, so the
// per-plane stores are contiguous across threads (coalesced b16 stores).
__global__ __launch_bounds__(256) void kan_build_act(
    const float* __restrict__ x, const float* __restrict__ grid,
    _Float16* __restrict__ A) {
  int idx = blockIdx.x * blockDim.x + threadIdx.x;   // [0, BATCH*IN_DIM)
  int m = idx / IN_DIM;
  int i = idx % IN_DIM;
  float xv = x[idx];
  float s  = xv / (1.0f + __expf(-xv));              // silu

  const float* g = grid + i * NKNOT;
  float gl[NKNOT];
#pragma unroll
  for (int j = 0; j < NKNOT; ++j) gl[j] = g[j];

  // order-0 indicator bases (11), then Cox–de Boor up to order 3 (8 bases)
  float b[NKNOT - 1];
#pragma unroll
  for (int j = 0; j < NKNOT - 1; ++j)
    b[j] = (xv >= gl[j] && xv < gl[j + 1]) ? 1.0f : 0.0f;
#pragma unroll
  for (int k = 1; k <= 3; ++k) {
#pragma unroll
    for (int j = 0; j < NKNOT - 1 - 3; ++j) {        // only first (11-k) matter;
      if (j < NKNOT - 1 - k) {                       // bound made constant-friendly
        float left  = (xv - gl[j])         / (gl[j + k]     - gl[j]);
        float right = (gl[j + k + 1] - xv) / (gl[j + k + 1] - gl[j + 1]);
        b[j] = left * b[j] + right * b[j + 1];
      }
    }
#pragma unroll
    for (int j = NKNOT - 1 - 3; j < NKNOT - 2; ++j) {
      if (j < NKNOT - 1 - k) {
        float left  = (xv - gl[j])         / (gl[j + k]     - gl[j]);
        float right = (gl[j + k + 1] - xv) / (gl[j + k + 1] - gl[j + 1]);
        b[j] = left * b[j] + right * b[j + 1];
      }
    }
  }

  _Float16* arow = A + (size_t)m * KDIM;
  arow[i] = (_Float16)s;                              // plane 0: silu
#pragma unroll
  for (int t = 0; t < NCOEF; ++t)                     // planes 1..8: bases
    arow[(size_t)(1 + t) * IN_DIM + i] = (_Float16)b[t];
}

// --------------------------- Kernel 2: build W -----------------------------
// One thread per (out n, input i). Same 9-plane K layout as A.
__global__ __launch_bounds__(256) void kan_build_w(
    const float* __restrict__ bw, const float* __restrict__ sw,
    const float* __restrict__ sc, _Float16* __restrict__ W) {
  int idx = blockIdx.x * blockDim.x + threadIdx.x;   // [0, OUT_DIM*IN_DIM)
  int i = idx % IN_DIM;
  int n = idx / IN_DIM;
  float scale = sc[idx];
  _Float16* wrow = W + (size_t)n * KDIM;
  wrow[i] = (_Float16)bw[idx];
#pragma unroll
  for (int t = 0; t < NCOEF; ++t)
    wrow[(size_t)(1 + t) * IN_DIM + i] =
        (_Float16)(sw[(size_t)idx * NCOEF + t] * scale);
}

// --------------------------- Kernel 3: WMMA GEMM ---------------------------
// out[m][n] = sum_k A[m][k] * W[n][k].  Block 128 threads = 4 waves (wave32),
// block tile 128x128, wave tile 64x64 = 4x4 of 16x16 WMMA tiles.
__global__ __launch_bounds__(128) void kan_wmma_gemm(
    const _Float16* __restrict__ A, const _Float16* __restrict__ W,
    float* __restrict__ out) {
  const int lane = threadIdx.x & 31;
  const int wave = threadIdx.x >> 5;
  const int wm = (wave >> 1) * 64;          // wave M offset in block tile
  const int wn = (wave & 1) * 64;           // wave N offset in block tile
  const int bm = blockIdx.x * 128;          // block M origin
  const int bn = blockIdx.y * 128;          // block N origin
  const int l16   = lane & 15;
  const int khalf = lane >> 4;              // 0 or 1

  v8f acc[4][4] = {};

  // per-lane base pointers (row for A fragments, column-row of W for B frags)
  const _Float16* arow[4];
  const _Float16* brow[4];
#pragma unroll
  for (int t = 0; t < 4; ++t) {
    arow[t] = A + (size_t)(bm + wm + t * 16 + l16) * KDIM + khalf * 8;
    brow[t] = W + (size_t)(bn + wn + t * 16 + l16) * KDIM + khalf * 8;
  }

  for (int k0 = 0; k0 < KDIM; k0 += BK) {
    v16h afrag[4], bfrag[4];
#pragma unroll
    for (int t = 0; t < 4; ++t) {
      // A 16x32 f16 fragment: halfs[0..7] = K[k0+khalf*8 ..], halfs[8..15] = +16
      union { v16h v; v8h h[2]; } ua;
      ua.h[0] = *(const v8h*)(arow[t] + k0);
      ua.h[1] = *(const v8h*)(arow[t] + k0 + 16);
      afrag[t] = ua.v;
      // B 32x16 f16 fragment: halfs[0..7] = K[k0+khalf*8 ..], halfs[8..15] = +16
      union { v16h v; v8h h[2]; } ub;
      ub.h[0] = *(const v8h*)(brow[t] + k0);
      ub.h[1] = *(const v8h*)(brow[t] + k0 + 16);
      bfrag[t] = ub.v;
    }
#pragma unroll
    for (int tm = 0; tm < 4; ++tm)
#pragma unroll
      for (int tn = 0; tn < 4; ++tn)
        acc[tm][tn] = __builtin_amdgcn_wmma_f32_16x16x32_f16(
            /*neg_a=*/false, afrag[tm], /*neg_b=*/false, bfrag[tn],
            /*c_mod=*/(short)0, acc[tm][tn],
            /*reuse_a=*/false, /*reuse_b=*/false);
  }

  // C/D layout: VGPR r -> row r + (lane/16)*8, col lane%16
#pragma unroll
  for (int tm = 0; tm < 4; ++tm) {
#pragma unroll
    for (int tn = 0; tn < 4; ++tn) {
      int ncol = bn + wn + tn * 16 + l16;
#pragma unroll
      for (int r = 0; r < 8; ++r) {
        int mrow = bm + wm + tm * 16 + r + khalf * 8;
        out[(size_t)mrow * OUT_DIM + ncol] = acc[tm][tn][r];
      }
    }
  }
}

// ------------------------------- launcher ----------------------------------
extern "C" void kernel_launch(void* const* d_in, const int* in_sizes, int n_in,
                              void* d_out, int out_size, void* d_ws, size_t ws_size,
                              hipStream_t stream) {
  const float* x    = (const float*)d_in[0];   // (4096,1024)
  const float* bw   = (const float*)d_in[1];   // (1024,1024)
  const float* sw   = (const float*)d_in[2];   // (1024,1024,8)
  const float* sc   = (const float*)d_in[3];   // (1024,1024)
  const float* grid = (const float*)d_in[4];   // (1024,12)

  _Float16* A = (_Float16*)d_ws;                                   // 75.5 MB
  _Float16* W = (_Float16*)((char*)d_ws + (size_t)BATCH * KDIM * sizeof(_Float16));
  float* out = (float*)d_out;

  kan_build_act<<<(BATCH * IN_DIM) / 256, 256, 0, stream>>>(x, grid, A);
  kan_build_w<<<(OUT_DIM * IN_DIM) / 256, 256, 0, stream>>>(bw, sw, sc, W);

  dim3 g(BATCH / 128, OUT_DIM / 128);          // (32, 8)
  kan_wmma_gemm<<<g, 128, 0, stream>>>(A, W, out);
}
